// MultiHeadAttention_36292473651979
// MI455X (gfx1250) — compile-verified, matrix operand order checked by reading, not run on hardware
//
#include <hip/hip_runtime.h>
#include <hip/hip_bf16.h>
#include <stdint.h>

#define B_  2
#define S_  2048
#define H_  32
#define D_  128
#define DM_ 4096
#define MTOT_ (B_ * S_)          // 4096 rows for the projection GEMMs
#define SCALEF 0.08838834764831845f   // 128^-0.5

typedef __attribute__((ext_vector_type(16))) __bf16 v16bf;
typedef __attribute__((ext_vector_type(8)))  __bf16 v8bf;
typedef __attribute__((ext_vector_type(8)))  float  v8f;

union FragBF {
    v16bf v;
    v8bf  h[2];
    uint4 q[2];
};

__device__ __forceinline__ unsigned short f2bf(float f) {
    union { float f; uint32_t u; } c; c.f = f;
    uint32_t u = c.u;
    return (unsigned short)((u + 0x7fffu + ((u >> 16) & 1u)) >> 16);
}

__device__ __forceinline__ v8f wmma_bf16(const FragBF& a, const FragBF& b, v8f c) {
    return __builtin_amdgcn_wmma_f32_16x16x32_bf16(
        /*neg_a=*/false, a.v, /*neg_b=*/false, b.v,
        /*c_mod=*/(short)0, c, /*reuse_a=*/false, /*reuse_b=*/false);
}

// ---------------------------------------------------------------------------
// CDNA5 async copy: global -> LDS without a VGPR round trip (ASYNCcnt).
// GVS mode: mem_addr = SGPR64 base + VGPR32 byte offset. VDST holds the LDS
// byte address; for the LDS aperture the low 32 bits of a generic pointer
// ARE the LDS address (ISA 10.2).
// ---------------------------------------------------------------------------
__device__ __forceinline__ void async_b128_to_lds(uint32_t ldsAddr,
                                                  const void* sbase,
                                                  uint32_t vByteOff) {
    asm volatile("global_load_async_to_lds_b128 %0, %1, %2"
                 :: "v"(ldsAddr), "v"(vByteOff), "s"(sbase)
                 : "memory");
}
__device__ __forceinline__ void wait_asynccnt0() {
    asm volatile("s_wait_asynccnt 0" ::: "memory");
}

// ---------------------------------------------------------------------------
// GEMM: Out = X @ W^T.  X: [4096,4096] (fp32 or bf16), W: [4096,4096] fp32.
// HEADS_OUT: write bf16 to [B,H,S,D]; else fp32 row-major [M,N].
// Block: 256 thr = 8 waves (2 M-waves x 4 N-waves); 128x256 tile; KT=64.
// Each wave owns a 64x64 sub-tile (4x4 WMMA tiles): per 32-K step,
// 8 fragment loads feed 16 WMMAs (ds:wmma = 1.0).
// ---------------------------------------------------------------------------
template <bool XBF16, bool HEADS_OUT>
__global__ __launch_bounds__(256)
void gemm_xwt(const void* __restrict__ Xraw,
              const float* __restrict__ W,
              void* __restrict__ Out)
{
    constexpr int KT   = 64;
    constexpr int LSTR = 72;   // shorts; 144B row stride: 16B aligned, conflict-free
    __shared__ __align__(16) unsigned short Xs[128 * LSTR];   // 18.4 KB
    __shared__ __align__(16) unsigned short Ws[256 * LSTR];   // 36.9 KB

    const int tid  = threadIdx.x;
    const int lane = tid & 31;
    const int wave = tid >> 5;
    const int wm   = wave >> 2;            // 0..1  (64 M rows each)
    const int wn   = wave & 3;             // 0..3  (64 N cols each)
    const int m0   = blockIdx.y * 128;
    const int n0   = blockIdx.x * 256;

    const int r15 = lane & 15;
    const int kb  = (lane >> 4) * 8;       // K-half select per A/B layout

    v8f acc[4][4];
#pragma unroll
    for (int mi = 0; mi < 4; ++mi)
#pragma unroll
        for (int ni = 0; ni < 4; ++ni)
            acc[mi][ni] = v8f{0.f,0.f,0.f,0.f,0.f,0.f,0.f,0.f};

    for (int k0 = 0; k0 < DM_; k0 += KT) {
        // ---- stage X tile (128 x 64) into LDS as bf16 ----
        if (XBF16) {
            // async global->LDS copy (data already bf16, no transform needed)
            const unsigned short* X = (const unsigned short*)Xraw;
            const unsigned short* base = X + (size_t)m0 * DM_ + k0;  // uniform
#pragma unroll
            for (int it = 0; it < 4; ++it) {
                int idx = tid + it * 256;            // 0..1023
                int row = idx >> 3;                  // 8 x 8-elem chunks per row
                int c8  = idx & 7;
                uint32_t ldsAddr = (uint32_t)(uintptr_t)&Xs[row * LSTR + c8 * 8];
                uint32_t off     = (uint32_t)(row * DM_ + c8 * 8) * 2u;
                async_b128_to_lds(ldsAddr, base, off);
            }
        } else {
            const float* X = (const float*)Xraw;
#pragma unroll
            for (int it = 0; it < 8; ++it) {
                int idx = tid + it * 256;            // 0..2047
                int row = idx >> 4;                  // 16 x float4 per row
                int c4  = idx & 15;
                float4 d = *(const float4*)(X + (size_t)(m0 + row) * DM_ + k0 + c4 * 4);
                union { unsigned short s[4]; uint2 u; } pk;
                pk.s[0] = f2bf(d.x); pk.s[1] = f2bf(d.y);
                pk.s[2] = f2bf(d.z); pk.s[3] = f2bf(d.w);
                *(uint2*)(&Xs[row * LSTR + c4 * 4]) = pk.u;
            }
        }
        // ---- stage W tile (256 rows of W, this K slice) ----
#pragma unroll
        for (int it = 0; it < 16; ++it) {
            int idx = tid + it * 256;                // 0..4095
            int row = idx >> 4;                      // 0..255
            int c4  = idx & 15;
            float4 d = *(const float4*)(W + (size_t)(n0 + row) * DM_ + k0 + c4 * 4);
            union { unsigned short s[4]; uint2 u; } pk;
            pk.s[0] = f2bf(d.x); pk.s[1] = f2bf(d.y);
            pk.s[2] = f2bf(d.z); pk.s[3] = f2bf(d.w);
            *(uint2*)(&Ws[row * LSTR + c4 * 4]) = pk.u;
        }
        // prefetch next K slice (emits global_prefetch_b8)
        if (k0 + KT < DM_) {
            __builtin_prefetch((const char*)W + ((size_t)(n0 + (tid >> 1)) * DM_ + k0 + KT) * 4, 0, 0);
        }
        if (XBF16) {
            wait_asynccnt0();                        // drain this wave's async copies
        }
        __syncthreads();

#pragma unroll
        for (int kc = 0; kc < 2; ++kc) {             // two 32-deep WMMA steps
            FragBF a[4], b[4];
#pragma unroll
            for (int mi = 0; mi < 4; ++mi) {
                const unsigned short* p = &Xs[(wm * 64 + mi * 16 + r15) * LSTR + kc * 32];
                a[mi].h[0] = *(const v8bf*)(p + kb);
                a[mi].h[1] = *(const v8bf*)(p + 16 + kb);
            }
#pragma unroll
            for (int ni = 0; ni < 4; ++ni) {
                const unsigned short* p = &Ws[(wn * 64 + ni * 16 + r15) * LSTR + kc * 32];
                b[ni].h[0] = *(const v8bf*)(p + kb);
                b[ni].h[1] = *(const v8bf*)(p + 16 + kb);
            }
#pragma unroll
            for (int mi = 0; mi < 4; ++mi)
#pragma unroll
                for (int ni = 0; ni < 4; ++ni)
                    acc[mi][ni] = wmma_bf16(a[mi], b[ni], acc[mi][ni]);
        }
        __syncthreads();
    }

    // ---- store ----
#pragma unroll
    for (int mi = 0; mi < 4; ++mi) {
#pragma unroll
        for (int ni = 0; ni < 4; ++ni) {
#pragma unroll
            for (int i = 0; i < 8; ++i) {
                int m = m0 + wm * 64 + mi * 16 + i + ((lane >> 4) << 3);
                int n = n0 + wn * 64 + ni * 16 + (lane & 15);
                float val = acc[mi][ni][i];
                if (HEADS_OUT) {
                    unsigned short* o = (unsigned short*)Out;
                    int b = m >> 11;          // / S_
                    int s = m & (S_ - 1);
                    int h = n >> 7;           // / D_
                    int d = n & (D_ - 1);
                    o[(((size_t)(b * H_ + h)) * S_ + s) * D_ + d] = f2bf(val);
                } else {
                    float* o = (float*)Out;
                    o[(size_t)m * DM_ + n] = val;
                }
            }
        }
    }
}

// ---------------------------------------------------------------------------
// Flash attention: one block per (b,h, 128 Q rows). 8 waves x 16 Q rows each.
// KV processed in blocks of 64. Q fragments live in registers; K staged with
// async global->LDS copies; V transposed through VGPRs so PV fragments are
// contiguous 16B LDS loads.
// ---------------------------------------------------------------------------
__global__ __launch_bounds__(256)
void attn_kernel(const unsigned short* __restrict__ Qh,
                 const unsigned short* __restrict__ Kh,
                 const unsigned short* __restrict__ Vh,
                 unsigned short* __restrict__ Ctx)
{
    constexpr int KSTR = 136;  // 272B: 16B aligned, conflict-free
    constexpr int VSTR = 72;   // 144B
    constexpr int PSTR = 72;
    __shared__ __align__(16) unsigned short Ks[64 * KSTR];      // K block [64][128]
    __shared__ __align__(16) unsigned short Vt[128 * VSTR];     // V block transposed [128][64]
    __shared__ __align__(16) unsigned short Ps[8 * 16 * PSTR];  // per-wave P scratch

    const int tid  = threadIdx.x;
    const int lane = tid & 31;
    const int wave = tid >> 5;
    const int q0   = blockIdx.x * 128;
    const int bh   = blockIdx.y;
    const int b    = bh / H_;
    const int h    = bh % H_;
    const size_t headOff = (size_t)bh * S_ * D_;

    const int r15 = lane & 15;
    const int kb  = (lane >> 4) * 8;

    // preload Q fragments (global bf16 is already in A-frag friendly layout)
    FragBF qf[4];
    {
        const unsigned short* qrow = Qh + headOff + (size_t)(q0 + wave * 16 + r15) * D_;
#pragma unroll
        for (int j = 0; j < 4; ++j) {
            qf[j].q[0] = *(const uint4*)(qrow + j * 32 + kb);
            qf[j].q[1] = *(const uint4*)(qrow + j * 32 + 16 + kb);
        }
    }

    v8f o[8];
#pragma unroll
    for (int no = 0; no < 8; ++no)
        o[no] = v8f{0.f,0.f,0.f,0.f,0.f,0.f,0.f,0.f};
    float rm[8], rl[8];
#pragma unroll
    for (int i = 0; i < 8; ++i) { rm[i] = -3.0e38f; rl[i] = 0.f; }

    for (int kv0 = 0; kv0 < S_; kv0 += 64) {
        // ---- stage K block via async copy: 64 rows x 16 chunks = 1024 slots ----
        {
            const unsigned short* kbase = Kh + headOff + (size_t)kv0 * D_;   // uniform
#pragma unroll
            for (int it = 0; it < 4; ++it) {
                int idx = tid + it * 256;          // 0..1023
                int s   = idx >> 4;                // 0..63
                int d8  = idx & 15;                // chunk of 8 bf16
                uint32_t ldsAddr = (uint32_t)(uintptr_t)&Ks[s * KSTR + d8 * 8];
                uint32_t off     = (uint32_t)(s * D_ + d8 * 8) * 2u;
                async_b128_to_lds(ldsAddr, kbase, off);
            }
        }
        // ---- stage V block, transposed (through VGPRs) ----
#pragma unroll
        for (int it = 0; it < 4; ++it) {
            int idx = tid + it * 256;          // 0..1023
            int s   = idx >> 4;                // 0..63
            int d8  = idx & 15;
            union { uint4 u; unsigned short sh[8]; } d;
            d.u = *(const uint4*)(Vh + headOff + (size_t)(kv0 + s) * D_ + d8 * 8);
#pragma unroll
            for (int j = 0; j < 8; ++j)
                Vt[(d8 * 8 + j) * VSTR + s] = d.sh[j];
        }
        wait_asynccnt0();                      // drain this wave's async K copies
        __syncthreads();

        // ---- scores: S = Q @ K^T  (16 x 64 per wave) ----
        v8f sc[4];
#pragma unroll
        for (int ni = 0; ni < 4; ++ni)
            sc[ni] = v8f{0.f,0.f,0.f,0.f,0.f,0.f,0.f,0.f};
#pragma unroll
        for (int j = 0; j < 4; ++j) {
#pragma unroll
            for (int ni = 0; ni < 4; ++ni) {
                FragBF kf;
                const unsigned short* p = &Ks[(ni * 16 + r15) * KSTR + j * 32];
                kf.h[0] = *(const v8bf*)(p + kb);
                kf.h[1] = *(const v8bf*)(p + 16 + kb);
                sc[ni] = wmma_bf16(qf[j], kf, sc[ni]);
            }
        }
#pragma unroll
        for (int ni = 0; ni < 4; ++ni)
#pragma unroll
            for (int i = 0; i < 8; ++i)
                sc[ni][i] *= SCALEF;

        // ---- online softmax (row r lives in VGPR r%8, lane-half r/8) ----
#pragma unroll
        for (int i = 0; i < 8; ++i) {
            float mx = fmaxf(fmaxf(sc[0][i], sc[1][i]), fmaxf(sc[2][i], sc[3][i]));
#pragma unroll
            for (int off = 8; off >= 1; off >>= 1)
                mx = fmaxf(mx, __shfl_xor(mx, off, 32));
            float mnew = fmaxf(rm[i], mx);
            float corr = __expf(rm[i] - mnew);
            rm[i] = mnew;
            float rs = 0.f;
#pragma unroll
            for (int ni = 0; ni < 4; ++ni) {
                float p = __expf(sc[ni][i] - mnew);
                sc[ni][i] = p;
                rs += p;
            }
#pragma unroll
            for (int off = 8; off >= 1; off >>= 1)
                rs += __shfl_xor(rs, off, 32);
            rl[i] = rl[i] * corr + rs;
#pragma unroll
            for (int no = 0; no < 8; ++no)
                o[no][i] *= corr;
        }

        // ---- re-layout P (C-layout -> A-layout) via wave-private LDS ----
        unsigned short* pw = &Ps[wave * 16 * PSTR];
#pragma unroll
        for (int ni = 0; ni < 4; ++ni)
#pragma unroll
            for (int i = 0; i < 8; ++i) {
                int row = i + ((lane >> 4) << 3);
                int col = ni * 16 + (lane & 15);
                pw[row * PSTR + col] = f2bf(sc[ni][i]);
            }

        // ---- O += P @ V ----
#pragma unroll
        for (int kc = 0; kc < 2; ++kc) {
            FragBF pf;
            const unsigned short* pp = &pw[r15 * PSTR + kc * 32];
            pf.h[0] = *(const v8bf*)(pp + kb);
            pf.h[1] = *(const v8bf*)(pp + 16 + kb);
#pragma unroll
            for (int no = 0; no < 8; ++no) {
                FragBF vf;
                const unsigned short* vp = &Vt[(no * 16 + r15) * VSTR + kc * 32];
                vf.h[0] = *(const v8bf*)(vp + kb);
                vf.h[1] = *(const v8bf*)(vp + 16 + kb);
                o[no] = wmma_bf16(pf, vf, o[no]);
            }
        }
        __syncthreads();
    }

    // ---- epilogue: normalize and write ctx in [B,S,DM] bf16 ----
#pragma unroll
    for (int i = 0; i < 8; ++i) {
        float inv = 1.0f / rl[i];
        int srow = q0 + wave * 16 + i + ((lane >> 4) << 3);
        size_t base = ((size_t)b * S_ + srow) * DM_ + (size_t)h * D_;
#pragma unroll
        for (int no = 0; no < 8; ++no) {
            int d = no * 16 + (lane & 15);
            Ctx[base + d] = f2bf(o[no][i] * inv);
        }
    }
}

// ---------------------------------------------------------------------------
extern "C" void kernel_launch(void* const* d_in, const int* in_sizes, int n_in,
                              void* d_out, int out_size, void* d_ws, size_t ws_size,
                              hipStream_t stream)
{
    (void)in_sizes; (void)n_in; (void)out_size; (void)ws_size;
    const float* q  = (const float*)d_in[0];
    const float* k  = (const float*)d_in[1];
    const float* v  = (const float*)d_in[2];
    const float* Wq = (const float*)d_in[3];
    const float* Wk = (const float*)d_in[4];
    const float* Wv = (const float*)d_in[5];
    const float* Wo = (const float*)d_in[6];

    const size_t HEAD_ELEMS = (size_t)B_ * H_ * S_ * D_;   // 16.8M bf16 each
    unsigned short* qh  = (unsigned short*)d_ws;
    unsigned short* kh  = qh + HEAD_ELEMS;
    unsigned short* vh  = kh + HEAD_ELEMS;
    unsigned short* ctx = vh + HEAD_ELEMS;

    dim3 gg(DM_ / 256, MTOT_ / 128);        // (16, 32)
    gemm_xwt<false, true><<<gg, 256, 0, stream>>>(q, Wq, qh);
    gemm_xwt<false, true><<<gg, 256, 0, stream>>>(k, Wk, kh);
    gemm_xwt<false, true><<<gg, 256, 0, stream>>>(v, Wv, vh);

    dim3 ga(S_ / 128, B_ * H_);             // (16, 64)
    attn_kernel<<<ga, 256, 0, stream>>>(qh, kh, vh, ctx);

    gemm_xwt<true, false><<<gg, 256, 0, stream>>>(ctx, Wo, d_out);
}
